// ImprovedMultiheadDilatedAttention_61727269978795
// MI455X (gfx1250) — matrix operand order repaired and checked by reading, not compiled, not run on hardware
//
#include <hip/hip_runtime.h>

// ---------------------------------------------------------------------------
// Types for CDNA5 WMMA (wave32, v_wmma_f32_16x16x32_bf16)
// ---------------------------------------------------------------------------
typedef __attribute__((ext_vector_type(16))) __bf16 v16bf;
typedef __attribute__((ext_vector_type(8)))  float  v8f;
typedef __attribute__((ext_vector_type(4)))  unsigned int v4u;
typedef __attribute__((ext_vector_type(4)))  int v4i;

#define WMMA_BF16(A, B, C) \
    __builtin_amdgcn_wmma_f32_16x16x32_bf16(false, (A), false, (B), (short)0, (C), false, false)

// Async global->LDS staging (CDNA5 GLOBAL_LOAD_ASYNC_TO_LDS_B128, ASYNCcnt).
#if defined(__HIP_DEVICE_COMPILE__) && \
    __has_builtin(__builtin_amdgcn_global_load_async_to_lds_b128) && \
    __has_builtin(__builtin_amdgcn_s_wait_asynccnt)
#define USE_ASYNC_LDS 1
#else
#define USE_ASYNC_LDS 0
#endif

#if USE_ASYNC_LDS
static __device__ inline void async_cp16(const unsigned short* g, unsigned short* l) {
    __builtin_amdgcn_global_load_async_to_lds_b128(
        (__attribute__((address_space(1))) v4i*)(g),
        (__attribute__((address_space(3))) v4i*)(l), 0, 0);
}
#endif

// fp32 -> bf16 (round to nearest even), storage type unsigned short
static __device__ inline unsigned short f2bf(float f) {
    union { float f; unsigned u; } x; x.f = f;
    unsigned r = x.u + 0x7FFFu + ((x.u >> 16) & 1u);
    return (unsigned short)(r >> 16);
}

// ---------------------------------------------------------------------------
// Fragment loaders (ISA 7.12.2 layouts). Storage in LDS as unsigned short,
// rows 16B-aligned (pitch in elements must be multiple of 8).
// A 16x32 (MxK): lane m = l&15; K-halves {0,8}+{0..7} and 16+{0,8}+{0..7}.
// B 32x16 (KxN) from a tile whose ROWS are the N dim, cols the K dim:
//   lane n = l&15; same K order as A: {0,8}+{0..7} and 16+{0,8}+{0..7}.
// ---------------------------------------------------------------------------
static __device__ inline v16bf load_frag_a(const unsigned short* base, int pitch) {
    int lane = threadIdx.x & 31;
    int m  = lane & 15;
    int kb = (lane >> 4) << 3;           // 0 or 8
    const unsigned short* p = base + m * pitch;
    union { v16bf b; v4u u[2]; } f;
    f.u[0] = *(const v4u*)(p + kb);       // K = kb .. kb+7
    f.u[1] = *(const v4u*)(p + 16 + kb);  // K = 16+kb .. 16+kb+7
    return f.b;
}

static __device__ inline v16bf load_frag_b(const unsigned short* base, int pitch) {
    int lane = threadIdx.x & 31;
    int n  = lane & 15;
    int kb = (lane >> 4) << 3;           // 0 or 8
    const unsigned short* p = base + n * pitch + kb;
    union { v16bf b; v4u u[2]; } f;
    f.u[0] = *(const v4u*)(p);            // K = kb .. kb+7
    f.u[1] = *(const v4u*)(p + 16);       // K = 16+kb .. 16+kb+7
    return f.b;
}

// ===========================================================================
// Kernel 1: fused QKV projection.  out = X @ Wqkv[sel]^T + b, q scaled 0.125.
// Writes bf16 q/k/v in (B, H, N, 64) layout.  Grid: (36 col-tiles, 128 row-tiles)
// ===========================================================================
__global__ __launch_bounds__(256) void qkv_gemm(
    const float* __restrict__ Xq, const float* __restrict__ Xk,
    const float* __restrict__ Xv, const float* __restrict__ W,
    const float* __restrict__ bias,
    unsigned short* __restrict__ q_ws, unsigned short* __restrict__ k_ws,
    unsigned short* __restrict__ v_ws)
{
    __shared__ unsigned short As[128 * 40];   // 128 x 32, pitch 40 (80B rows)
    __shared__ unsigned short Bs[64 * 40];    // 64 x 32

    const int tid  = threadIdx.x;
    const int lane = tid & 31;
    const int w    = tid >> 5;
    const int colBase = blockIdx.x * 64;      // 0..2303
    const int mBase   = blockIdx.y * 128;     // 0..16383
    const int sel = colBase / 768;            // 0=q 1=k 2=v
    const float* X = (sel == 0) ? Xq : ((sel == 1) ? Xk : Xv);
    const int wm = w >> 1, wn = w & 1;        // wave tile: 32(M) x 32(N)

    v8f acc[2][2] = {};

    for (int kt = 0; kt < 24; ++kt) {
        int k0 = kt * 32;
        __syncthreads();
#pragma unroll
        for (int i = 0; i < 16; ++i) {
            int idx = tid + i * 256; int r = idx >> 5; int c = idx & 31;
            As[r * 40 + c] = f2bf(X[(size_t)(mBase + r) * 768 + k0 + c]);
        }
#pragma unroll
        for (int i = 0; i < 8; ++i) {
            int idx = tid + i * 256; int r = idx >> 5; int c = idx & 31;
            Bs[r * 40 + c] = f2bf(W[(size_t)(colBase + r) * 768 + k0 + c]);
        }
        __syncthreads();

        v16bf a0 = load_frag_a(As + (wm * 32) * 40, 40);
        v16bf a1 = load_frag_a(As + (wm * 32 + 16) * 40, 40);
        v16bf b0 = load_frag_b(Bs + (wn * 32) * 40, 40);
        v16bf b1 = load_frag_b(Bs + (wn * 32 + 16) * 40, 40);
        acc[0][0] = WMMA_BF16(a0, b0, acc[0][0]);
        acc[0][1] = WMMA_BF16(a0, b1, acc[0][1]);
        acc[1][0] = WMMA_BF16(a1, b0, acc[1][0]);
        acc[1][1] = WMMA_BF16(a1, b1, acc[1][1]);
    }

    const int colb = lane & 15;
    const int ha   = (lane >> 4) << 3;        // rows j (lanes<16) or j+8
    unsigned short* dst = (sel == 0) ? q_ws : ((sel == 1) ? k_ws : v_ws);
    const float qs = (sel == 0) ? 0.125f : 1.0f;   // 1/sqrt(64) baked into q

#pragma unroll
    for (int mt = 0; mt < 2; ++mt)
#pragma unroll
        for (int nt = 0; nt < 2; ++nt)
#pragma unroll
            for (int j = 0; j < 8; ++j) {
                int grow = mBase + wm * 32 + mt * 16 + j + ha;
                int gcol = colBase + wn * 32 + nt * 16 + colb;
                float v = (acc[mt][nt][j] + bias[gcol]) * qs;
                int cc = gcol - sel * 768;
                int hh = cc >> 6, dd = cc & 63;
                int bb = grow >> 13, nn = grow & 8191;
                dst[((size_t)(bb * 12 + hh) * 8192 + nn) * 64 + dd] = f2bf(v);
            }
}

// ===========================================================================
// Kernel 2: dilated flash attention.  t = 2048 dilated tokens for ALL groups.
// Block = 128 queries x (loop 32 x 64 keys).  8 waves, each owns 16 q rows.
// Q/K tiles staged with GLOBAL_LOAD_ASYNC_TO_LDS_B128 (ASYNCcnt); V staged
// through VGPRs because it lands transposed in LDS.
// Grid: (16 q-blocks, 2*mSegs*4 instances).
// ===========================================================================
__global__ __launch_bounds__(256) void dilated_attn(
    const unsigned short* __restrict__ q_ws, const unsigned short* __restrict__ k_ws,
    const unsigned short* __restrict__ v_ws, float* __restrict__ attn_out,
    const int* __restrict__ flags,
    int headBase, int segLen, int rate, int off, int mSegs)
{
    __shared__ unsigned short Qs [128 * 72];  // 128 x 64, pitch 72 (144B rows)
    __shared__ unsigned short Ks [64 * 72];   // keys x d (row-major)
    __shared__ unsigned short Vts[64 * 72];   // d x keys (transposed V)
    __shared__ unsigned short Ps [128 * 72];  // per-wave 16-row P tiles

    const int tid  = threadIdx.x;
    const int lane = tid & 31;
    const int w    = tid >> 5;
    const int colb = lane & 15;
    const int ha   = (lane >> 4) << 3;
    const int causal = flags[0];

    const int idxY = blockIdx.y;
    const int b    = idxY / (mSegs * 4);
    const int rem  = idxY % (mSegs * 4);
    const int seg  = rem >> 2;
    const int h    = headBase + (rem & 3);
    const size_t rowbase = ((size_t)b * 12 + h) * 8192;
    const int segBase = seg * segLen + off;
    const int qtok0 = blockIdx.x * 128;
    const int wrow  = w * 16;

    // ---- stage Q (128 x 64 bf16) and keep A-fragments in registers --------
#if USE_ASYNC_LDS
#pragma unroll
    for (int i = 0; i < 4; ++i) {              // 1024 16B chunks (128 rows x 8)
        int idx = tid + i * 256; int r = idx >> 3; int c = idx & 7;
        int n = segBase + (qtok0 + r) * rate;
        async_cp16(q_ws + (rowbase + n) * 64 + c * 8, Qs + r * 72 + c * 8);
    }
    __builtin_amdgcn_s_wait_asynccnt(0);
#else
#pragma unroll
    for (int i = 0; i < 16; ++i) {
        int idx = tid + i * 256; int r = idx >> 5; int c = idx & 31;
        int n = segBase + (qtok0 + r) * rate;
        ((unsigned int*)(Qs + r * 72))[c] =
            ((const unsigned int*)(q_ws + (rowbase + n) * 64))[c];
    }
#endif
    __syncthreads();
    v16bf aq0 = load_frag_a(Qs + wrow * 72, 72);
    v16bf aq1 = load_frag_a(Qs + wrow * 72 + 32, 72);

    float mrow[8], lrow[8];
    v8f oacc[4] = {};
#pragma unroll
    for (int j = 0; j < 8; ++j) { mrow[j] = -1e30f; lrow[j] = 0.f; }

    for (int it = 0; it < 32; ++it) {
        const int kb64 = it * 64;
        __syncthreads();                       // protect Ks/Vts reuse
        // ---- stage K (row-major, async DMA) ------------------------------
#if USE_ASYNC_LDS
#pragma unroll
        for (int i = 0; i < 2; ++i) {          // 512 16B chunks (64 rows x 8)
            int idx = tid + i * 256; int kr = idx >> 3; int c = idx & 7;
            int n = segBase + (kb64 + kr) * rate;
            async_cp16(k_ws + (rowbase + n) * 64 + c * 8, Ks + kr * 72 + c * 8);
        }
#else
#pragma unroll
        for (int i = 0; i < 8; ++i) {
            int idx = tid + i * 256; int kr = idx >> 5; int c = idx & 31;
            int n = segBase + (kb64 + kr) * rate;
            ((unsigned int*)(Ks + kr * 72))[c] =
                ((const unsigned int*)(k_ws + (rowbase + n) * 64))[c];
        }
#endif
        // ---- stage V transposed (VGPR path, must transpose) --------------
#pragma unroll
        for (int i = 0; i < 8; ++i) {
            int idx = tid + i * 256; int kr = idx >> 5; int c = idx & 31;
            int n = segBase + (kb64 + kr) * rate;
            unsigned int pv = ((const unsigned int*)(v_ws + (rowbase + n) * 64))[c];
            Vts[(2 * c) * 72 + kr]     = (unsigned short)(pv & 0xFFFFu);
            Vts[(2 * c + 1) * 72 + kr] = (unsigned short)(pv >> 16);
            if (it + 1 < 32) {                 // prefetch next key block
                int n2 = segBase + (kb64 + 64 + kr) * rate;
                __builtin_prefetch(k_ws + (rowbase + n2) * 64, 0, 1);
                __builtin_prefetch(v_ws + (rowbase + n2) * 64, 0, 1);
            }
        }
#if USE_ASYNC_LDS
        __builtin_amdgcn_s_wait_asynccnt(0);
#endif
        __syncthreads();

        // ---- S = Q K^T  (4 col tiles x 2 K-steps) ------------------------
        v8f s[4] = {};
#pragma unroll
        for (int nt = 0; nt < 4; ++nt) {
            v16bf bk0 = load_frag_b(Ks + nt * 16 * 72, 72);
            v16bf bk1 = load_frag_b(Ks + nt * 16 * 72 + 32, 72);
            s[nt] = WMMA_BF16(aq0, bk0, s[nt]);
            s[nt] = WMMA_BF16(aq1, bk1, s[nt]);
        }

        // ---- in-wave online softmax (rows live in-wave) ------------------
#pragma unroll
        for (int j = 0; j < 8; ++j) {
            const int jr = qtok0 + wrow + j + ha;
            float mx = -1e30f;
#pragma unroll
            for (int nt = 0; nt < 4; ++nt) {
                float sv = s[nt][j];
                if (causal && (kb64 + nt * 16 + colb > jr)) sv = -1e30f;
                mx = fmaxf(mx, sv);
            }
            mx = fmaxf(mx, __shfl_xor(mx, 1, 32));
            mx = fmaxf(mx, __shfl_xor(mx, 2, 32));
            mx = fmaxf(mx, __shfl_xor(mx, 4, 32));
            mx = fmaxf(mx, __shfl_xor(mx, 8, 32));
            float mnew  = fmaxf(mrow[j], mx);
            float alpha = __expf(mrow[j] - mnew);
            mrow[j] = mnew;
            lrow[j] *= alpha;
#pragma unroll
            for (int dt = 0; dt < 4; ++dt) oacc[dt][j] *= alpha;

            float rs = 0.f;
#pragma unroll
            for (int nt = 0; nt < 4; ++nt) {
                bool msk = causal && (kb64 + nt * 16 + colb > jr);
                float p = msk ? 0.f : __expf(s[nt][j] - mnew);
                rs += p;
                Ps[(wrow + j + ha) * 72 + nt * 16 + colb] = f2bf(p);
            }
            rs += __shfl_xor(rs, 1, 32);
            rs += __shfl_xor(rs, 2, 32);
            rs += __shfl_xor(rs, 4, 32);
            rs += __shfl_xor(rs, 8, 32);
            lrow[j] += rs;
        }
        asm volatile("s_wait_dscnt 0" ::: "memory");   // P stores -> P loads

        // ---- O += P V  (reload P as A-fragments, Vts rows are B) ---------
        v16bf ap0 = load_frag_a(Ps + wrow * 72, 72);
        v16bf ap1 = load_frag_a(Ps + wrow * 72 + 32, 72);
#pragma unroll
        for (int dt = 0; dt < 4; ++dt) {
            v16bf bv0 = load_frag_b(Vts + dt * 16 * 72, 72);
            v16bf bv1 = load_frag_b(Vts + dt * 16 * 72 + 32, 72);
            oacc[dt] = WMMA_BF16(ap0, bv0, oacc[dt]);
            oacc[dt] = WMMA_BF16(ap1, bv1, oacc[dt]);
        }
    }

    // ---- epilogue: normalize, scatter into (B, N, H, 64) fp32 ------------
#pragma unroll
    for (int j = 0; j < 8; ++j) {
        float inv = 1.0f / fmaxf(lrow[j], 1e-20f);
        int jr = qtok0 + wrow + j + ha;
        int n  = segBase + jr * rate;
#pragma unroll
        for (int dt = 0; dt < 4; ++dt) {
            int d = dt * 16 + colb;
            attn_out[(((size_t)b * 8192 + n) * 12 + h) * 64 + d] = oacc[dt][j] * inv;
        }
    }
}

// ===========================================================================
// Kernel 3: row LayerNorm (768) -> bf16.  Grid: 16384 blocks x 256 threads.
// ===========================================================================
__global__ __launch_bounds__(256) void ln_norm(
    const float* __restrict__ x, const float* __restrict__ lnw,
    const float* __restrict__ lnb, unsigned short* __restrict__ xn)
{
    __shared__ float red[16];
    const int row = blockIdx.x;
    const int tid = threadIdx.x;
    const float* xr = x + (size_t)row * 768;
    float v0 = xr[tid], v1 = xr[tid + 256], v2 = xr[tid + 512];
    float s = v0 + v1 + v2;
    float q = v0 * v0 + v1 * v1 + v2 * v2;
#pragma unroll
    for (int m = 1; m < 32; m <<= 1) {
        s += __shfl_xor(s, m, 32);
        q += __shfl_xor(q, m, 32);
    }
    if ((tid & 31) == 0) { red[tid >> 5] = s; red[8 + (tid >> 5)] = q; }
    __syncthreads();
    float ts = 0.f, tq = 0.f;
#pragma unroll
    for (int i = 0; i < 8; ++i) { ts += red[i]; tq += red[8 + i]; }
    float mu  = ts * (1.f / 768.f);
    float var = tq * (1.f / 768.f) - mu * mu;
    float rs  = rsqrtf(var + 1e-5f);
    unsigned short* xo = xn + (size_t)row * 768;
    xo[tid]       = f2bf((v0 - mu) * rs * lnw[tid]       + lnb[tid]);
    xo[tid + 256] = f2bf((v1 - mu) * rs * lnw[tid + 256] + lnb[tid + 256]);
    xo[tid + 512] = f2bf((v2 - mu) * rs * lnw[tid + 512] + lnb[tid + 512]);
}

// ===========================================================================
// Kernel 4: output projection  out = xn @ Wo^T + bo  (fp32 out)
// Grid: (12 col-tiles, 128 row-tiles)
// ===========================================================================
__global__ __launch_bounds__(256) void oproj_gemm(
    const unsigned short* __restrict__ xn, const float* __restrict__ W,
    const float* __restrict__ bias, float* __restrict__ out)
{
    __shared__ unsigned short As[128 * 40];
    __shared__ unsigned short Bs[64 * 40];

    const int tid  = threadIdx.x;
    const int lane = tid & 31;
    const int w    = tid >> 5;
    const int colBase = blockIdx.x * 64;
    const int mBase   = blockIdx.y * 128;
    const int wm = w >> 1, wn = w & 1;

    v8f acc[2][2] = {};

    for (int kt = 0; kt < 24; ++kt) {
        int k0 = kt * 32;
        __syncthreads();
#pragma unroll
        for (int i = 0; i < 8; ++i) {          // bf16 A: copy 128x16 dwords
            int idx = tid + i * 256; int r = idx >> 4; int c = idx & 15;
            ((unsigned int*)(As + r * 40))[c] =
                ((const unsigned int*)(xn + (size_t)(mBase + r) * 768 + k0))[c];
        }
#pragma unroll
        for (int i = 0; i < 8; ++i) {          // fp32 W -> bf16
            int idx = tid + i * 256; int r = idx >> 5; int c = idx & 31;
            Bs[r * 40 + c] = f2bf(W[(size_t)(colBase + r) * 768 + k0 + c]);
        }
        __syncthreads();

        v16bf a0 = load_frag_a(As + (wm * 32) * 40, 40);
        v16bf a1 = load_frag_a(As + (wm * 32 + 16) * 40, 40);
        v16bf b0 = load_frag_b(Bs + (wn * 32) * 40, 40);
        v16bf b1 = load_frag_b(Bs + (wn * 32 + 16) * 40, 40);
        acc[0][0] = WMMA_BF16(a0, b0, acc[0][0]);
        acc[0][1] = WMMA_BF16(a0, b1, acc[0][1]);
        acc[1][0] = WMMA_BF16(a1, b0, acc[1][0]);
        acc[1][1] = WMMA_BF16(a1, b1, acc[1][1]);
    }

    const int colb = lane & 15;
    const int ha   = (lane >> 4) << 3;
#pragma unroll
    for (int mt = 0; mt < 2; ++mt)
#pragma unroll
        for (int nt = 0; nt < 2; ++nt)
#pragma unroll
            for (int j = 0; j < 8; ++j) {
                int grow = mBase + wm * 32 + mt * 16 + j + ha;
                int gcol = colBase + wn * 32 + nt * 16 + colb;
                out[(size_t)grow * 768 + gcol] = acc[mt][nt][j] + bias[gcol];
            }
}

// ===========================================================================
// Host launch
// ===========================================================================
extern "C" void kernel_launch(void* const* d_in, const int* in_sizes, int n_in,
                              void* d_out, int out_size, void* d_ws, size_t ws_size,
                              hipStream_t stream)
{
    (void)in_sizes; (void)n_in; (void)out_size; (void)ws_size;
    const float* Xq   = (const float*)d_in[0];
    const float* Xk   = (const float*)d_in[1];
    const float* Xv   = (const float*)d_in[2];
    const float* Wqkv = (const float*)d_in[3];
    const float* bqkv = (const float*)d_in[4];
    const float* Wo   = (const float*)d_in[5];
    const float* bo   = (const float*)d_in[6];
    const float* lnw  = (const float*)d_in[7];
    const float* lnb  = (const float*)d_in[8];
    const int*   flg  = (const int*)d_in[9];
    float* out = (float*)d_out;

    char* ws = (char*)d_ws;
    const size_t qkvElems = (size_t)2 * 12 * 8192 * 64;       // per tensor
    unsigned short* q_ws = (unsigned short*)ws;  ws += qkvElems * 2;
    unsigned short* k_ws = (unsigned short*)ws;  ws += qkvElems * 2;
    unsigned short* v_ws = (unsigned short*)ws;  ws += qkvElems * 2;
    float* attn_ws = (float*)ws;                 ws += (size_t)16384 * 768 * 4;
    unsigned short* xn_ws = (unsigned short*)ws;

    // 1) fused QKV projection (bf16 WMMA, q pre-scaled by 1/sqrt(64))
    qkv_gemm<<<dim3(36, 128), 256, 0, stream>>>(Xq, Xk, Xv, Wqkv, bqkv,
                                                q_ws, k_ws, v_ws);
    // 2) zero the attention output (dilation leaves structural zeros)
    (void)hipMemsetAsync(attn_ws, 0, (size_t)16384 * 768 * 4, stream);
    // 3) dilated flash attention, one launch per (segment, dilation) group
    dilated_attn<<<dim3(16, 32), 256, 0, stream>>>(q_ws, k_ws, v_ws, attn_ws,
                                                   flg, 0, 2048, 1, 0, 4);
    dilated_attn<<<dim3(16, 16), 256, 0, stream>>>(q_ws, k_ws, v_ws, attn_ws,
                                                   flg, 4, 4096, 2, 1, 2);
    dilated_attn<<<dim3(16, 8), 256, 0, stream>>>(q_ws, k_ws, v_ws, attn_ws,
                                                  flg, 8, 8192, 4, 2, 1);
    // 4) LayerNorm -> bf16
    ln_norm<<<16384, 256, 0, stream>>>(attn_ws, lnw, lnb, xn_ws);
    // 5) output projection -> fp32
    oproj_gemm<<<dim3(12, 128), 256, 0, stream>>>(xn_ws, Wo, bo, out);
}
